// SINKAttention_16140487098656
// MI455X (gfx1250) — matrix operand order repaired and checked by reading, not compiled, not run on hardware
//
#include <hip/hip_runtime.h>
#include <stdint.h>

#define L_SEQ    2048
#define HEAD_DIM 32
#define NUM_HEAD 8
#define BATCH    2
#define BHN      (BATCH * NUM_HEAD)     // 16
#define QTILES   (L_SEQ / 16)           // 128
#define CHUNKS   (L_SEQ / 32)           // 64
#define WPB      4                      // waves per block (wave32)
#define TPB      (WPB * 32)
#define MASKVAL  (-10000.0f)
#define SCALE    0.17677669529663687f   // 1/sqrt(32)
#define LOG_L    7.6246189861593985f    // log(2048)
#define LOG2E    1.4426950408889634f
#define LN2      0.6931471805599453f

typedef uint32_t u32;
typedef __attribute__((ext_vector_type(16))) __bf16    v16bf;
typedef __attribute__((ext_vector_type(8)))  float     v8f;
typedef __attribute__((ext_vector_type(4)))  uint32_t  v4u;

union BFrag { alignas(16) uint32_t u[8]; v4u q[2]; v16bf v; };

__device__ __forceinline__ float fexp2(float x) { return __builtin_amdgcn_exp2f(x); }   // v_exp_f32
__device__ __forceinline__ float flog2(float x) { return __builtin_amdgcn_logf(x); }    // v_log_f32

// hardware cvt: pack two fp32 -> packed bf16 dword (RNE), optional residuals
__device__ __forceinline__ u32 cvt_pk_r(float x0, float x1, float& r0, float& r1) {
    union { __bf16 b[2]; u32 u; } t;
    t.b[0] = (__bf16)x0; t.b[1] = (__bf16)x1;
    r0 = x0 - (float)t.b[0];
    r1 = x1 - (float)t.b[1];
    return t.u;
}
__device__ __forceinline__ u32 cvt_pk(float x0, float x1) {
    union { __bf16 b[2]; u32 u; } t;
    t.b[0] = (__bf16)x0; t.b[1] = (__bf16)x1;
    return t.u;
}

// Same-wave LDS RAW/WAR: DS pipe is in-order per wave; wait DScnt + block compiler motion.
__device__ __forceinline__ void lds_fence() {
    asm volatile("s_wait_dscnt 0" ::: "memory");
}

// C += A*B with 3-term hi/lo split (near-fp32 accuracy on bf16 WMMA).
__device__ __forceinline__ v8f wmma3(const BFrag& ah, const BFrag& al,
                                     const BFrag& bh, const BFrag& bl, v8f c) {
    c = __builtin_amdgcn_wmma_f32_16x16x32_bf16(false, ah.v, false, bh.v, (short)0, c, false, false);
    c = __builtin_amdgcn_wmma_f32_16x16x32_bf16(false, ah.v, false, bl.v, (short)0, c, false, false);
    c = __builtin_amdgcn_wmma_f32_16x16x32_bf16(false, al.v, false, bh.v, (short)0, c, false, false);
    return c;
}

// Hoisted-pointer fragment fetches: base points at this lane's first 16B block of row 0.
// token-tile t is 16 rows * 16 dwords = 64 v4u away per tile.
__device__ __forceinline__ void fetch_b(const v4u* pb, int o, BFrag& f) {  // B-frag: q[0]=+0, q[1]=+1
    f.q[0] = pb[o];
    f.q[1] = pb[o + 1];
}
__device__ __forceinline__ void fetch_a(const v4u* pa, int o, BFrag& f) {  // A-frag: q[0]=+0, q[1]=+1
    f.q[0] = pa[o];
    f.q[1] = pa[o + 1];
}

// ---------------- k0a: convert Q*(scale*log2e) and K to packed bf16 hi/lo ----------------
__global__ __launch_bounds__(TPB) void k0_cvt_qk(const float* __restrict__ Q,
                                                 const float* __restrict__ K,
                                                 u32* __restrict__ Qh, u32* __restrict__ Ql,
                                                 u32* __restrict__ Kh, u32* __restrict__ Kl) {
    const size_t row = (size_t)blockIdx.x * TPB + threadIdx.x;   // token row, 0..BHN*L-1
    const float qs = SCALE * LOG2E;   // WMMA then yields S*log2e, consumed by v_exp_f32 (=2^x)
    const float4* q4 = (const float4*)(Q + row * HEAD_DIM);
    const float4* k4 = (const float4*)(K + row * HEAD_DIM);
    alignas(16) u32 qh[16], ql[16], kh[16], kl[16];
#pragma unroll
    for (int i = 0; i < 8; ++i) {
        float4 tq = q4[i], tk = k4[i];
        float r0, r1;
        qh[2 * i]     = cvt_pk_r(tq.x * qs, tq.y * qs, r0, r1);
        ql[2 * i]     = cvt_pk(r0, r1);
        qh[2 * i + 1] = cvt_pk_r(tq.z * qs, tq.w * qs, r0, r1);
        ql[2 * i + 1] = cvt_pk(r0, r1);
        kh[2 * i]     = cvt_pk_r(tk.x, tk.y, r0, r1);
        kl[2 * i]     = cvt_pk(r0, r1);
        kh[2 * i + 1] = cvt_pk_r(tk.z, tk.w, r0, r1);
        kl[2 * i + 1] = cvt_pk(r0, r1);
    }
#pragma unroll
    for (int i = 0; i < 4; ++i) {
        ((v4u*)(Qh + row * 16))[i] = *(v4u*)&qh[4 * i];
        ((v4u*)(Ql + row * 16))[i] = *(v4u*)&ql[4 * i];
        ((v4u*)(Kh + row * 16))[i] = *(v4u*)&kh[4 * i];
        ((v4u*)(Kl + row * 16))[i] = *(v4u*)&kl[4 * i];
    }
}

// ---------------- k0b: V -> V^T (feature-major) packed bf16 hi/lo ----------------
__global__ __launch_bounds__(TPB) void k0_cvt_vT(const float* __restrict__ V,
                                                 u32* __restrict__ Vth, u32* __restrict__ Vtl) {
    __shared__ float Vst[WPB][32][32];
    const int wid = threadIdx.x >> 5;
    const int lane = threadIdx.x & 31;
    const int gw = blockIdx.x * WPB + wid;     // tile id, 0..BHN*64-1
    const int bh = gw >> 6;
    const int kb = (gw & 63) * 32;

    const float4* vp = (const float4*)(V + ((size_t)bh * L_SEQ + kb + lane) * HEAD_DIM);
    float4* dst = (float4*)&Vst[wid][lane][0];
#pragma unroll
    for (int j = 0; j < 8; ++j) dst[j] = vp[j];
    lds_fence();

    const int d = lane;                        // feature this lane transposes
    alignas(16) u32 hv[16], lv[16];
#pragma unroll
    for (int j = 0; j < 16; ++j) {
        float x0 = Vst[wid][2 * j][d], x1 = Vst[wid][2 * j + 1][d];
        float r0, r1;
        hv[j] = cvt_pk_r(x0, x1, r0, r1);
        lv[j] = cvt_pk(r0, r1);
    }
    u32* oh = Vth + ((size_t)bh * 32 + d) * (L_SEQ / 2) + kb / 2;
    u32* ol = Vtl + ((size_t)bh * 32 + d) * (L_SEQ / 2) + kb / 2;
#pragma unroll
    for (int i = 0; i < 4; ++i) {
        ((v4u*)oh)[i] = *(v4u*)&hv[4 * i];
        ((v4u*)ol)[i] = *(v4u*)&lv[4 * i];
    }
}

// ---------------- Kernel 1: w_q = exp(u_q) = 1 / (L * sum_j exp(S_qj_masked)) ----------------
__global__ __launch_bounds__(TPB) void k1_row_lse(const u32* __restrict__ Qh, const u32* __restrict__ Ql,
                                                  const u32* __restrict__ Kh, const u32* __restrict__ Kl,
                                                  const int* __restrict__ mask, float* __restrict__ w) {
    __shared__ float mb[L_SEQ];
    __shared__ float red[WPB][16][16];
    const int wid = threadIdx.x >> 5;
    const int lane = threadIdx.x & 31;
    const int l16 = lane & 15;
    const int h = lane >> 4;
    const int gw = blockIdx.x * WPB + wid;
    const int bh = gw >> 7;
    const int qt = gw & (QTILES - 1);
    const int b = bh >> 3;

    for (int i = threadIdx.x; i < L_SEQ; i += TPB)
        mb[i] = mask[b * L_SEQ + i] ? (MASKVAL * LOG2E) : 0.0f;   // log2-domain bias
    __syncthreads();

    const size_t tokbase = (size_t)bh * L_SEQ;
    BFrag qhf, qlf;
    fetch_a((const v4u*)(Qh + (tokbase + qt * 16 + l16) * 16) + 2 * h, 0, qhf);
    fetch_a((const v4u*)(Ql + (tokbase + qt * 16 + l16) * 16) + 2 * h, 0, qlf);

    // loop-invariant per-lane base pointers (B-frag of K^T)
    const v4u* pKh = (const v4u*)(Kh + (tokbase + l16) * 16) + 2 * h;
    const v4u* pKl = (const v4u*)(Kl + (tokbase + l16) * 16) + 2 * h;

    float acc[8] = {};
#pragma unroll 2
    for (int kt = 0; kt < QTILES; kt += 2) {
        BFrag k0h, k0l, k1h, k1l;
        fetch_b(pKh, kt * 64,      k0h);
        fetch_b(pKl, kt * 64,      k0l);
        fetch_b(pKh, kt * 64 + 64, k1h);
        fetch_b(pKl, kt * 64 + 64, k1l);
        v8f s0 = {}; s0 = wmma3(qhf, qlf, k0h, k0l, s0);   // S*log2e
        v8f s1 = {}; s1 = wmma3(qhf, qlf, k1h, k1l, s1);
        float mv0 = mb[kt * 16 + l16];
        float mv1 = mb[kt * 16 + 16 + l16];
        float e[16];
#pragma unroll
        for (int r = 0; r < 8; ++r) { e[r] = s0[r] + mv0; e[8 + r] = s1[r] + mv1; }
#pragma unroll
        for (int i = 0; i < 16; ++i) e[i] = fexp2(e[i]);
#pragma unroll
        for (int r = 0; r < 8; ++r) acc[r] += e[r] + e[8 + r];
    }
#pragma unroll
    for (int r = 0; r < 8; ++r) red[wid][r + 8 * h][l16] = acc[r];
    lds_fence();
    if (lane < 16) {
        float t = 0.0f;
#pragma unroll
        for (int n = 0; n < 16; ++n) t += red[wid][lane][n];
        t = fmaxf(t, 1e-37f);
        w[bh * L_SEQ + qt * 16 + lane] = 1.0f / ((float)L_SEQ * t);   // exp(u)
    }
}

// ---------------- Kernel 2: v_j = -logL - LSE_q(S_masked + u_q) ----------------
__global__ __launch_bounds__(TPB) void k2_col_lse(const u32* __restrict__ Qh, const u32* __restrict__ Ql,
                                                  const u32* __restrict__ Kh, const u32* __restrict__ Kl,
                                                  const int* __restrict__ mask, const float* __restrict__ w,
                                                  float* __restrict__ v) {
    __shared__ float wlds[L_SEQ];
    __shared__ float red[WPB][32];
    __shared__ float wred[WPB][32];
    const int wid = threadIdx.x >> 5;
    const int lane = threadIdx.x & 31;
    const int l16 = lane & 15;
    const int h = lane >> 4;
    const int gw = blockIdx.x * WPB + wid;
    const int bh = gw >> 7;
    const int kt = gw & (QTILES - 1);
    const int b = bh >> 3;

    for (int i = threadIdx.x; i < L_SEQ; i += TPB)
        wlds[i] = w[bh * L_SEQ + i];
    __syncthreads();

    const size_t tokbase = (size_t)bh * L_SEQ;
    BFrag khf, klf;
    fetch_b((const v4u*)(Kh + (tokbase + kt * 16 + l16) * 16) + 2 * h, 0, khf);
    fetch_b((const v4u*)(Kl + (tokbase + kt * 16 + l16) * 16) + 2 * h, 0, klf);
    const float madd = mask[b * L_SEQ + kt * 16 + l16] ? (MASKVAL * LOG2E) : 0.0f;

    const v4u* pQh = (const v4u*)(Qh + (tokbase + l16) * 16) + 2 * h;
    const v4u* pQl = (const v4u*)(Ql + (tokbase + l16) * 16) + 2 * h;

    float acc = 0.0f;
#pragma unroll 2
    for (int qt = 0; qt < QTILES; qt += 2) {
        BFrag q0h, q0l, q1h, q1l;
        fetch_a(pQh, qt * 64,      q0h);
        fetch_a(pQl, qt * 64,      q0l);
        fetch_a(pQh, qt * 64 + 64, q1h);
        fetch_a(pQl, qt * 64 + 64, q1l);
        v8f s0 = {}; s0 = wmma3(q0h, q0l, khf, klf, s0);   // S*log2e
        v8f s1 = {}; s1 = wmma3(q1h, q1l, khf, klf, s1);
        float e[16];
#pragma unroll
        for (int r = 0; r < 8; ++r) { e[r] = s0[r] + madd; e[8 + r] = s1[r] + madd; }
#pragma unroll
        for (int i = 0; i < 16; ++i) e[i] = fexp2(e[i]);
#pragma unroll
        for (int r = 0; r < 8; ++r) {
            acc = fmaf(wlds[qt * 16 + r + 8 * h],      e[r],     acc);
            acc = fmaf(wlds[qt * 16 + 16 + r + 8 * h], e[8 + r], acc);
        }
    }
    // tail-only: wt = sum_q exp(u_q) (needed for masked-column stable LSE)
    float ws = 0.0f;
    for (int i = lane; i < L_SEQ; i += 32) ws += wlds[i];
    red[wid][lane] = acc;
    wred[wid][lane] = ws;
    lds_fence();
    if (lane < 16) {
        float t = red[wid][lane] + red[wid][lane + 16];
        float wt = 0.0f;
#pragma unroll
        for (int n = 0; n < 32; ++n) wt += wred[wid][n];
        int mk = mask[b * L_SEQ + kt * 16 + lane];
        float vj = mk ? (-LOG_L - (MASKVAL + flog2(fmaxf(wt, 1e-37f)) * LN2))
                      : (-LOG_L - flog2(fmaxf(t, 1e-37f)) * LN2);
        v[bh * L_SEQ + kt * 16 + lane] = vj;
    }
}

// ---------------- Kernel 3: out = softmax_j(a_j*S + b_j) @ V ----------------
__global__ __launch_bounds__(TPB) void k3_out(const u32* __restrict__ Qh, const u32* __restrict__ Ql,
                                              const u32* __restrict__ Kh, const u32* __restrict__ Kl,
                                              const u32* __restrict__ Vth, const u32* __restrict__ Vtl,
                                              const int* __restrict__ mask, const float* __restrict__ v,
                                              float* __restrict__ out) {
    __shared__ float ka[L_SEQ];            // 1 (keep S) or 0 (masked)
    __shared__ float kb[L_SEQ];            // (v_j + mask*MASKVAL) * log2e
    __shared__ float Pst[WPB][16][32];     // per-wave P transpose stage
    __shared__ float dred[WPB][16][16];
    __shared__ float dtot[WPB][16];
    const int wid = threadIdx.x >> 5;
    const int lane = threadIdx.x & 31;
    const int l16 = lane & 15;
    const int h = lane >> 4;
    const int gw = blockIdx.x * WPB + wid;
    const int bh = gw >> 7;
    const int qt = gw & (QTILES - 1);
    const int b = bh >> 3;

    for (int i = threadIdx.x; i < L_SEQ; i += TPB) {
        int mk = mask[b * L_SEQ + i];
        ka[i] = mk ? 0.0f : 1.0f;
        kb[i] = (v[bh * L_SEQ + i] + (mk ? MASKVAL : 0.0f)) * LOG2E;
    }
    __syncthreads();

    const size_t tokbase = (size_t)bh * L_SEQ;
    BFrag qhf, qlf;
    fetch_a((const v4u*)(Qh + (tokbase + qt * 16 + l16) * 16) + 2 * h, 0, qhf);
    fetch_a((const v4u*)(Ql + (tokbase + qt * 16 + l16) * 16) + 2 * h, 0, qlf);

    // loop-invariant per-lane bases
    const v4u* pKh = (const v4u*)(Kh + (tokbase + l16) * 16) + 2 * h;
    const v4u* pKl = (const v4u*)(Kl + (tokbase + l16) * 16) + 2 * h;
    const v4u* pV0h = (const v4u*)(Vth + ((size_t)bh * 32 + l16) * (L_SEQ / 2)) + 2 * h;
    const v4u* pV0l = (const v4u*)(Vtl + ((size_t)bh * 32 + l16) * (L_SEQ / 2)) + 2 * h;
    const v4u* pV1h = pV0h + 16 * (L_SEQ / 8);   // +16 feature rows (L/2 dwords = L/8 v4u each)
    const v4u* pV1l = pV0l + 16 * (L_SEQ / 8);

    v8f acc0 = {}, acc1 = {};
    float den[8] = {};

    for (int ch = 0; ch < CHUNKS; ++ch) {
        BFrag b0h, b0l, b1h, b1l;
        fetch_b(pKh, ch * 128,      b0h);       // keys ch*32 .. +15
        fetch_b(pKl, ch * 128,      b0l);
        fetch_b(pKh, ch * 128 + 64, b1h);       // keys ch*32+16 .. +31
        fetch_b(pKl, ch * 128 + 64, b1l);
        v8f s0 = {}; s0 = wmma3(qhf, qlf, b0h, b0l, s0);   // S*log2e
        v8f s1 = {}; s1 = wmma3(qhf, qlf, b1h, b1l, s1);

        // V^T B-fragments: chunk ch -> v4u offset ch*4 within each feature row
        BFrag v0h, v0l, v1h, v1l;
        fetch_b(pV0h, ch * 4, v0h);
        fetch_b(pV0l, ch * 4, v0l);
        fetch_b(pV1h, ch * 4, v1h);
        fetch_b(pV1l, ch * 4, v1l);

        const int kbase = ch * 32;
        const float a0 = ka[kbase + l16],      c0 = kb[kbase + l16];
        const float a1 = ka[kbase + 16 + l16], c1 = kb[kbase + 16 + l16];
        float e[16];
#pragma unroll
        for (int r = 0; r < 8; ++r) { e[r] = fmaf(s0[r], a0, c0); e[8 + r] = fmaf(s1[r], a1, c1); }
#pragma unroll
        for (int i = 0; i < 16; ++i) e[i] = fexp2(e[i]);   // exp(S_masked + v_j)
#pragma unroll
        for (int r = 0; r < 8; ++r) {
            den[r] += e[r] + e[8 + r];
            Pst[wid][r + 8 * h][l16] = e[r];
            Pst[wid][r + 8 * h][16 + l16] = e[8 + r];
        }
        lds_fence();

        // P: C-layout -> A-layout (16x32) via LDS, hardware cvt split
        BFrag pfh, pfl;
        {
            const int f0 = h * 16;
#pragma unroll
            for (int g = 0; g < 2; ++g)
#pragma unroll
                for (int j = 0; j < 4; ++j) {
                    int k0 = f0 + g * 8 + 2 * j;
                    float r0, r1;
                    pfh.u[g * 4 + j] = cvt_pk_r(Pst[wid][l16][k0], Pst[wid][l16][k0 + 1], r0, r1);
                    pfl.u[g * 4 + j] = cvt_pk(r0, r1);
                }
        }
        acc0 = wmma3(pfh, pfl, v0h, v0l, acc0);
        acc1 = wmma3(pfh, pfl, v1h, v1l, acc1);
    }

    // row denominators: 16-lane reduce, broadcast reciprocal
#pragma unroll
    for (int r = 0; r < 8; ++r) dred[wid][r + 8 * h][l16] = den[r];
    lds_fence();
    if (lane < 16) {
        float t = 0.0f;
#pragma unroll
        for (int n = 0; n < 16; ++n) t += dred[wid][lane][n];
        dtot[wid][lane] = 1.0f / fmaxf(t, 1e-37f);
    }
    lds_fence();

    float* op = out + ((size_t)bh * L_SEQ + qt * 16) * HEAD_DIM;
#pragma unroll
    for (int r = 0; r < 8; ++r) {
        float inv = dtot[wid][r + 8 * h];
        op[(r + 8 * h) * HEAD_DIM + l16]      = acc0[r] * inv;
        op[(r + 8 * h) * HEAD_DIM + 16 + l16] = acc1[r] * inv;
    }
}

extern "C" void kernel_launch(void* const* d_in, const int* in_sizes, int n_in,
                              void* d_out, int out_size, void* d_ws, size_t ws_size,
                              hipStream_t stream) {
    const float* Q = (const float*)d_in[0];
    const float* K = (const float*)d_in[1];
    const float* V = (const float*)d_in[2];
    const int* mask = (const int*)d_in[3];
    float* out = (float*)d_out;

    // workspace layout
    uint8_t* base = (uint8_t*)d_ws;
    const size_t NTOK = (size_t)BHN * L_SEQ;          // 32768
    float* w = (float*)base;                          // 128 KB
    float* v = (float*)(base + (NTOK * 4));           // 128 KB
    u32* Qh = (u32*)(base + 2 * NTOK * 4);            // each bf16 matrix: NTOK*16 dwords = 2 MB
    u32* Ql = Qh + NTOK * 16;
    u32* Kh = Ql + NTOK * 16;
    u32* Kl = Kh + NTOK * 16;
    u32* Vth = Kl + NTOK * 16;
    u32* Vtl = Vth + NTOK * 16;

    dim3 blk(TPB);
    k0_cvt_qk<<<dim3(NTOK / TPB), blk, 0, stream>>>(Q, K, Qh, Ql, Kh, Kl);            // 256 blocks
    k0_cvt_vT<<<dim3(BHN * (L_SEQ / 32) / WPB), blk, 0, stream>>>(V, Vth, Vtl);       // 256 blocks
    dim3 grd(BHN * QTILES / WPB);                                                      // 512 blocks
    k1_row_lse<<<grd, blk, 0, stream>>>(Qh, Ql, Kh, Kl, mask, w);
    k2_col_lse<<<grd, blk, 0, stream>>>(Qh, Ql, Kh, Kl, mask, w, v);
    k3_out<<<grd, blk, 0, stream>>>(Qh, Ql, Kh, Kl, Vth, Vtl, mask, v, out);
}
